// WindowAttentionBlock_10539849744586
// MI455X (gfx1250) — matrix operand order repaired and checked by reading, not compiled, not run on hardware
//
#include <hip/hip_runtime.h>

typedef __attribute__((ext_vector_type(16))) _Float16 v16h;
typedef __attribute__((ext_vector_type(8)))  _Float16 v8h;
typedef __attribute__((ext_vector_type(8)))  float    v8f;

namespace {

constexpr int C      = 384;
constexpr int TC     = 3 * C;      // 1152
constexpr int QK     = 2 * C;      // 768 (Q|K width in LDS)
constexpr int NHEAD  = 12;
constexpr int HD     = 32;         // head dim
constexpr int L      = 49;         // tokens per window
constexpr int LP     = 64;         // padded rows (4 M-tiles)
constexpr int WIN_N  = TC * C;     // 442368 elements of w_in
constexpr int WOUT_N = C * C;      // 147456 elements of w_out

// ---- WMMA fragment loaders, per CDNA5 ISA wave32 layouts ----

// A fragment, f16 16x32 (M x K). Lane l: M = l&15; lanes 0-15 hold
// K in {0..7, 16..23}, lanes 16-31 hold K in {8..15, 24..31}.
__device__ __forceinline__ v16h load_a(const _Float16* p, int ld, int lane) {
  const int m  = lane & 15;
  const int hi = lane >> 4;
  const _Float16* row = p + m * ld + hi * 8;
  v16h a;
#pragma unroll
  for (int j = 0; j < 8; ++j) a[j] = row[j];          // K = hi*8 + 0..7
#pragma unroll
  for (int j = 0; j < 8; ++j) a[8 + j] = row[16 + j]; // K = hi*8 + 16..23
  return a;
}

// B fragment, f16 32x16 (K x N), element (k,n) = p[n*ld + k]
// (source stores each output column n contiguously along k -> 2x b128/lane).
// Lane l: N = l&15; lanes 0-15 hold K in {0..7, 16..23}, lanes 16-31 hold K in {8..15, 24..31}.
__device__ __forceinline__ v16h load_bT(const _Float16* p, int ld, int lane) {
  const int n  = lane & 15;
  const int kb = (lane >> 4) << 3;
  const _Float16* q = p + n * ld + kb;
  v16h b;
#pragma unroll
  for (int i = 0; i < 16; ++i) b[i] = q[i < 8 ? i : i + 8];
  return b;
}

__device__ __forceinline__ v8f wmma16(v16h a, v16h b, v8f c) {
  // D = A x B + C, f32 accumulate
  return __builtin_amdgcn_wmma_f32_16x16x32_f16(false, a, false, b,
                                                (short)0, c, false, false);
}

} // namespace

// Convert weights (f32, L2-resident, reused by all 2048 blocks) to f16 in ws.
__global__ void swin_prep_weights(const float* __restrict__ w_in,
                                  const float* __restrict__ w_out,
                                  _Float16* __restrict__ wh) {
  int i = blockIdx.x * blockDim.x + threadIdx.x;
  if (i < WIN_N)  wh[i] = (_Float16)w_in[i];
  if (i < WOUT_N) wh[WIN_N + i] = (_Float16)w_out[i];
}

// One workgroup per 7x7 window. 256 threads = 8 wave32s.
__global__ __launch_bounds__(256, 1)
void swin_window_attn(const float* __restrict__ x,
                      const float* __restrict__ gamma,
                      const float* __restrict__ beta,
                      const float* __restrict__ b_in,
                      const float* __restrict__ b_out,
                      const _Float16* __restrict__ wh,   // [w_in f16 | w_out f16]
                      float* __restrict__ out) {
  extern __shared__ char smem_raw[];
  _Float16* sY  = (_Float16*)smem_raw;  // LP x C    LN output (f16)
  _Float16* sQK = sY + LP * C;          // LP x 768  Q | K (row-major)
  _Float16* sVt = sQK + LP * QK;        // C  x LP   V transposed: [ch][key]
  _Float16* sP  = sVt + C * LP;         // 8 waves x (16 x 64) softmax probs
  _Float16* sO  = sY;                   // attention output reuses sY

  const int tid  = threadIdx.x;
  const int lane = tid & 31;
  const int wv   = tid >> 5;
  const int hi   = lane >> 4;
  const int ln16 = lane & 15;

  const int wid = blockIdx.x;           // 2048 windows
  const int b0  = wid >> 6;
  const int rem = wid & 63;
  const int wh0 = rem >> 3;             // window row (0..7)
  const int ww0 = rem & 7;              // window col (0..7)

  // ---- Phase 1: LayerNorm -> sY (f16), zero-pad rows 49..63 ----
  for (int it = 0; it < 7; ++it) {      // uniform trip count
    const int l = wv + it * 8;
    if (l < L) {
      const int rr = l / 7, cc = l - rr * 7;
      const long tok = (long)(b0 * 56 + wh0 * 7 + rr) * 56 + (ww0 * 7 + cc);
      const float* xp = x + tok * (long)C;
      float xv[12], s = 0.f, sq = 0.f;
#pragma unroll
      for (int i = 0; i < 12; ++i) {
        float v = xp[lane + 32 * i];
        xv[i] = v; s += v; sq += v * v;
      }
#pragma unroll
      for (int m = 16; m >= 1; m >>= 1) {
        s  += __shfl_xor(s, m, 32);
        sq += __shfl_xor(sq, m, 32);
      }
      const float mu   = s * (1.f / C);
      const float rstd = rsqrtf(sq * (1.f / C) - mu * mu + 1e-5f);
#pragma unroll
      for (int i = 0; i < 12; ++i) {
        const int c = lane + 32 * i;
        sY[l * C + c] = (_Float16)((xv[i] - mu) * rstd * gamma[c] + beta[c]);
      }
    }
  }
  for (int i = tid; i < (LP - L) * C; i += 256) sY[L * C + i] = (_Float16)0.f;
  __syncthreads();

  // ---- Phase 2: QKV = Y @ w_in^T + b_in; Q scaled; V stored transposed ----
  // 144 N-tile-pairs (32 cols each); mt = (wv+8it)&3 is per-wave constant.
  for (int it = 0; it < 18; ++it) {
    const int q  = wv + it * 8;
    const int mt = q & 3;
    const int n0 = (q >> 2) << 5;       // pair base column (multiple of 32)
    const float bj0 = b_in[n0 + ln16];
    const float bj1 = b_in[n0 + 16 + ln16];
    v8f acc0 = {bj0, bj0, bj0, bj0, bj0, bj0, bj0, bj0};
    v8f acc1 = {bj1, bj1, bj1, bj1, bj1, bj1, bj1, bj1};
#pragma unroll 1
    for (int kc = 0; kc < C / 32; ++kc) {
      v16h a  = load_a(sY + mt * 16 * C + kc * 32, C, lane);
      v16h bA = load_bT(wh + (size_t)n0 * C + kc * 32, C, lane);
      v16h bB = load_bT(wh + (size_t)(n0 + 16) * C + kc * 32, C, lane);
      acc0 = wmma16(a, bA, acc0);
      acc1 = wmma16(a, bB, acc1);
    }
    if (n0 < QK) {                      // Q or K -> row-major sQK
      const float sc = (n0 < C) ? 0.17677669529663687f : 1.0f; // 1/sqrt(32)
#pragma unroll
      for (int r = 0; r < 8; ++r) {
        const int row = mt * 16 + r + hi * 8;
        sQK[row * QK + n0 + ln16]      = (_Float16)(acc0[r] * sc);
        sQK[row * QK + n0 + 16 + ln16] = (_Float16)(acc1[r] * sc);
      }
    } else {                            // V -> transposed sVt[ch][key]
      v8h h0, h1;
#pragma unroll
      for (int r = 0; r < 8; ++r) {
        h0[r] = (_Float16)acc0[r];
        h1[r] = (_Float16)acc1[r];
      }
      const int key0 = mt * 16 + hi * 8;           // 8 consecutive keys
      *(v8h*)(sVt + (n0 - QK + ln16) * LP + key0)      = h0;
      *(v8h*)(sVt + (n0 - QK + 16 + ln16) * LP + key0) = h1;
    }
  }
  __syncthreads();

  // ---- Phase 3: attention, task = (head, query m-tile), 48 tasks ----
  _Float16* sPw = sP + wv * (16 * 64);
  for (int it = 0; it < 6; ++it) {
    const int t    = wv + it * 8;
    const int head = t >> 2, mt = t & 3;
    v16h aq = load_a(sQK + mt * 16 * QK + head * HD, QK, lane);          // Q
    v8f S[4];
#pragma unroll
    for (int nt = 0; nt < 4; ++nt) {    // scores: K = HD = 32, 1 WMMA/tile
      v16h b = load_bT(sQK + nt * 16 * QK + C + head * HD, QK, lane);    // K^T
      v8f z = {0.f, 0.f, 0.f, 0.f, 0.f, 0.f, 0.f, 0.f};
      S[nt] = wmma16(aq, b, z);
    }
    // mask padded keys (col >= 49)
#pragma unroll
    for (int nt = 0; nt < 4; ++nt)
      if (nt * 16 + ln16 >= L) {
#pragma unroll
        for (int r = 0; r < 8; ++r) S[nt][r] = -3.0e38f;
      }
    // softmax: each matrix row lives in one 16-lane half (D layout)
#pragma unroll
    for (int r = 0; r < 8; ++r) {
      float mx = fmaxf(fmaxf(S[0][r], S[1][r]), fmaxf(S[2][r], S[3][r]));
#pragma unroll
      for (int m = 8; m >= 1; m >>= 1) mx = fmaxf(mx, __shfl_xor(mx, m, 32));
      float sum = 0.f;
#pragma unroll
      for (int nt = 0; nt < 4; ++nt) {
        float e = __expf(S[nt][r] - mx);
        S[nt][r] = e; sum += e;
      }
#pragma unroll
      for (int m = 8; m >= 1; m >>= 1) sum += __shfl_xor(sum, m, 32);
      const float inv = 1.f / sum;
#pragma unroll
      for (int nt = 0; nt < 4; ++nt) S[nt][r] *= inv;
    }
    // re-fragment P via per-wave LDS tile [16 x 64]
#pragma unroll
    for (int nt = 0; nt < 4; ++nt)
#pragma unroll
      for (int r = 0; r < 8; ++r)
        sPw[(r + hi * 8) * 64 + nt * 16 + ln16] = (_Float16)S[nt][r];
    // O = P @ V   (V transposed in LDS -> contiguous B fragments)
#pragma unroll
    for (int n2 = 0; n2 < 2; ++n2) {
      v8f acc = {0.f, 0.f, 0.f, 0.f, 0.f, 0.f, 0.f, 0.f};
#pragma unroll
      for (int kc = 0; kc < 2; ++kc) {
        v16h a = load_a(sPw + kc * 32, 64, lane);
        v16h b = load_bT(sVt + (head * HD + n2 * 16) * LP + kc * 32, LP, lane);
        acc = wmma16(a, b, acc);
      }
#pragma unroll
      for (int r = 0; r < 8; ++r)
        sO[(mt * 16 + r + hi * 8) * C + head * HD + n2 * 16 + ln16] =
            (_Float16)acc[r];
    }
  }
  __syncthreads();

  // ---- Phase 4: out = O @ w_out^T + b_out + shortcut, scatter ----
  const _Float16* gWout = wh + WIN_N;
  for (int it = 0; it < 6; ++it) {      // 48 N-tile-pairs
    const int q  = wv + it * 8;
    const int mt = q & 3;
    const int n0 = (q >> 2) << 5;
    const float bj0 = b_out[n0 + ln16];
    const float bj1 = b_out[n0 + 16 + ln16];
    v8f acc0 = {bj0, bj0, bj0, bj0, bj0, bj0, bj0, bj0};
    v8f acc1 = {bj1, bj1, bj1, bj1, bj1, bj1, bj1, bj1};
#pragma unroll 1
    for (int kc = 0; kc < C / 32; ++kc) {
      v16h a  = load_a(sO + mt * 16 * C + kc * 32, C, lane);
      v16h bA = load_bT(gWout + (size_t)n0 * C + kc * 32, C, lane);
      v16h bB = load_bT(gWout + (size_t)(n0 + 16) * C + kc * 32, C, lane);
      acc0 = wmma16(a, bA, acc0);
      acc1 = wmma16(a, bB, acc1);
    }
#pragma unroll
    for (int r = 0; r < 8; ++r) {
      const int m = mt * 16 + r + hi * 8;
      if (m < L) {
        const int rr = m / 7, cc = m - rr * 7;
        const long tok = (long)(b0 * 56 + wh0 * 7 + rr) * 56 + (ww0 * 7 + cc);
        const long idx = tok * (long)C + n0 + ln16;
        out[idx]      = x[idx] + acc0[r];
        out[idx + 16] = x[idx + 16] + acc1[r];
      }
    }
  }
}

extern "C" void kernel_launch(void* const* d_in, const int* in_sizes, int n_in,
                              void* d_out, int out_size, void* d_ws, size_t ws_size,
                              hipStream_t stream) {
  (void)in_sizes; (void)n_in; (void)out_size; (void)ws_size;
  const float* x     = (const float*)d_in[0];
  const float* gamma = (const float*)d_in[1];
  const float* beta  = (const float*)d_in[2];
  const float* w_in  = (const float*)d_in[3];
  const float* b_in  = (const float*)d_in[4];
  const float* w_out = (const float*)d_in[5];
  const float* b_out = (const float*)d_in[6];
  // d_in[7], d_in[8] are H, W (statically 56)
  float* out   = (float*)d_out;
  _Float16* wf = (_Float16*)d_ws;     // [WIN_N | WOUT_N] f16 weights

  swin_prep_weights<<<(WIN_N + 255) / 256, 256, 0, stream>>>(w_in, w_out, wf);

  const size_t smem =
      (size_t)(LP * C + LP * QK + C * LP + 8 * 16 * 64) * sizeof(_Float16);
  hipFuncSetAttribute(reinterpret_cast<const void*>(swin_window_attn),
                      hipFuncAttributeMaxDynamicSharedMemorySize, (int)smem);
  swin_window_attn<<<2048, 256, smem, stream>>>(x, gamma, beta, b_in, b_out,
                                                wf, out);
}
